// GCN_292057776415
// MI455X (gfx1250) — matrix operand order repaired and checked by reading, not compiled, not run on hardware
//
#include <hip/hip_runtime.h>
#include <hip/hip_bf16.h>

typedef __bf16 v16bf __attribute__((ext_vector_type(16)));
typedef float  v8f   __attribute__((ext_vector_type(8)));

#define NNODES 100000
#define NEDGES 1600000
#define NGRAPH 64
#define HDIM   128
#define NCLS   10
#define GN_EPS 1e-5f

__device__ __forceinline__ void atomAddF(float* p, float v) {
  __hip_atomic_fetch_add(p, v, __ATOMIC_RELAXED, __HIP_MEMORY_SCOPE_AGENT);
}

// ---------------- small utility kernels ----------------
__global__ void fill_f32(float* __restrict__ p, float v, int n) {
  int i = blockIdx.x * blockDim.x + threadIdx.x;
  if (i < n) p[i] = v;
}

__global__ void deg_edges(const int* __restrict__ col, float* __restrict__ deg, int E) {
  int i = blockIdx.x * blockDim.x + threadIdx.x;
  if (i < E) atomAddF(&deg[col[i]], 1.0f);
}

__global__ void deg_to_rsqrt(float* __restrict__ deg, int n) {
  int i = blockIdx.x * blockDim.x + threadIdx.x;
  if (i < n) { float d = deg[i]; deg[i] = (d > 0.f) ? rsqrtf(d) : 0.f; }
}

__global__ void count_nodes(const int* __restrict__ batch, float* __restrict__ counts, int n) {
  int i = blockIdx.x * blockDim.x + threadIdx.x;
  if (i < n) atomAddF(&counts[batch[i]], 1.0f);
}

// ---------------- weight pre-swizzle into WMMA B-fragment layout ----------------
// Fragment (kt in 0..3, nt in 0..7): 32 lanes x 16 bf16, contiguous per lane.
// Lane l<16 : col = 16*nt + l,      K = 32*kt + j       (j = 0..15)
// Lane l>=16: col = 16*nt + (l-16), K = 32*kt + 16 + j
__global__ void swizzle_w(const float* __restrict__ W, __bf16* __restrict__ Ws) {
  int idx = blockIdx.x * blockDim.x + threadIdx.x;       // 0 .. 16383
  if (idx >= HDIM * HDIM) return;
  int j    = idx & 15;
  int lane = (idx >> 4) & 31;
  int nt   = (idx >> 9) & 7;
  int kt   = idx >> 12;
  int k = kt * 32 + (lane >> 4) * 16 + j;
  int n = nt * 16 + (lane & 15);
  Ws[idx] = (__bf16)W[k * HDIM + n];
}

// ---------------- WMMA GEMM: out[N,128] = in[N,128] @ W[128,128] ----------------
// 256 threads = 8 waves; block covers 128 rows; wave w owns rows row0..row0+15.
// Swizzled weights staged once into LDS; all 4 A fragments preloaded/converted
// before the MMA loop so the WMMA stream only waits on dscnt (LDS B feeds).
__global__ __launch_bounds__(256) void gemm128_wmma(
    const float* __restrict__ in, const __bf16* __restrict__ Ws,
    float* __restrict__ out, int nrows) {
  __shared__ __bf16 sW[HDIM * HDIM];                   // 32 KB of 320 KB/WGP

  // cooperative stage: 2048 x b128 moves, 8 per thread
  {
    const int t = threadIdx.x;
    #pragma unroll
    for (int i = 0; i < 8; ++i) {
      int idx = t + i * 256;
      ((int4*)sW)[idx] = ((const int4*)Ws)[idx];
    }
  }

  const int lane = threadIdx.x & 31;
  const int wave = threadIdx.x >> 5;
  const int half = lane >> 4;
  const int lm   = lane & 15;
  const int row0 = blockIdx.x * 128 + wave * 16;

  int m = row0 + lm;
  if (m > nrows - 1) m = nrows - 1;                    // clamp loads, EXEC stays full
  const float* __restrict__ arow = in + (size_t)m * HDIM;

  // preload ALL A fragments (16 b128 loads, one clause) and convert to bf16
  v16bf afrag[4];
  #pragma unroll
  for (int kt = 0; kt < 4; ++kt) {
    const int k1 = kt * 32 + half * 16;
    const int k2 = kt * 32 + half * 16 + 8;
    const float4 p0 = *(const float4*)(arow + k1);
    const float4 p1 = *(const float4*)(arow + k1 + 4);
    const float4 p2 = *(const float4*)(arow + k2);
    const float4 p3 = *(const float4*)(arow + k2 + 4);
    afrag[kt][0]  = (__bf16)p0.x; afrag[kt][1]  = (__bf16)p0.y;
    afrag[kt][2]  = (__bf16)p0.z; afrag[kt][3]  = (__bf16)p0.w;
    afrag[kt][4]  = (__bf16)p1.x; afrag[kt][5]  = (__bf16)p1.y;
    afrag[kt][6]  = (__bf16)p1.z; afrag[kt][7]  = (__bf16)p1.w;
    afrag[kt][8]  = (__bf16)p2.x; afrag[kt][9]  = (__bf16)p2.y;
    afrag[kt][10] = (__bf16)p2.z; afrag[kt][11] = (__bf16)p2.w;
    afrag[kt][12] = (__bf16)p3.x; afrag[kt][13] = (__bf16)p3.y;
    afrag[kt][14] = (__bf16)p3.z; afrag[kt][15] = (__bf16)p3.w;
  }

  __syncthreads();                                     // weights staged

  const v8f vzero = {0.f, 0.f, 0.f, 0.f, 0.f, 0.f, 0.f, 0.f};
  v8f acc[8];
  #pragma unroll
  for (int nt = 0; nt < 8; ++nt) acc[nt] = vzero;

  #pragma unroll
  for (int kt = 0; kt < 4; ++kt) {
    #pragma unroll
    for (int nt = 0; nt < 8; ++nt) {
      v16bf b = *(const v16bf*)(sW + (((kt * 8 + nt) * 32) + lane) * 16);
      acc[nt] = __builtin_amdgcn_wmma_f32_16x16x32_bf16(
          false, afrag[kt], false, b, (short)0, acc[nt], false, false);
    }
  }

  // D layout: VGPR v -> row (row0 + v + 8*half), col = 16*nt + lm
  #pragma unroll
  for (int nt = 0; nt < 8; ++nt) {
    #pragma unroll
    for (int v = 0; v < 8; ++v) {
      int mr = row0 + v + 8 * half;
      if (mr < nrows) out[(size_t)mr * HDIM + nt * 16 + lm] = acc[nt][v];
    }
  }
}

// ---------------- sparse aggregation: agg[col] += h[row] * dis[row]*dis[col] ----------------
// one wave per edge; one b128 gather (4 features) per lane, 4 f32 L2 atomics
__global__ __launch_bounds__(256) void edge_scatter(
    const int* __restrict__ row, const int* __restrict__ col,
    const float* __restrict__ dis, const float* __restrict__ h,
    float* __restrict__ agg, int E) {
  const int lane = threadIdx.x & 31;
  const int e = blockIdx.x * 8 + (threadIdx.x >> 5);
  if (e >= E) return;
  const int r = row[e], c = col[e];
  const float nrm = dis[r] * dis[c];
  const float4 hv = *(const float4*)(h + (size_t)r * HDIM + lane * 4);
  float* __restrict__ ap = agg + (size_t)c * HDIM + lane * 4;
  atomAddF(ap + 0, hv.x * nrm);
  atomAddF(ap + 1, hv.y * nrm);
  atomAddF(ap + 2, hv.z * nrm);
  atomAddF(ap + 3, hv.w * nrm);
}

// ---------------- self-loop + bias + ReLU, accumulate GraphNorm mean ----------------
// z = relu(agg + h*dis^2 + b) -> written in place over h; msum[g] += z   (float4 lanes)
__global__ void node_epilogue(float* __restrict__ h, const float* __restrict__ agg,
    const float* __restrict__ dis, const float* __restrict__ bias,
    const int* __restrict__ batch, float* __restrict__ msum, int total4) {
  int idx = blockIdx.x * blockDim.x + threadIdx.x;
  if (idx >= total4) return;
  int n = idx >> 5, f4 = idx & 31;
  float d = dis[n], dd = d * d;
  float4 hv = ((const float4*)h)[idx];
  float4 av = ((const float4*)agg)[idx];
  float4 bv = ((const float4*)bias)[f4];
  float4 z;
  z.x = fmaxf(av.x + hv.x * dd + bv.x, 0.f);
  z.y = fmaxf(av.y + hv.y * dd + bv.y, 0.f);
  z.z = fmaxf(av.z + hv.z * dd + bv.z, 0.f);
  z.w = fmaxf(av.w + hv.w * dd + bv.w, 0.f);
  ((float4*)h)[idx] = z;
  float* ms = msum + (batch[n] << 7) + f4 * 4;
  atomAddF(ms + 0, z.x); atomAddF(ms + 1, z.y);
  atomAddF(ms + 2, z.z); atomAddF(ms + 3, z.w);
}

// mean = msum / counts (in place)
__global__ void finish_mean(float* __restrict__ msum, const float* __restrict__ counts, int total) {
  int i = blockIdx.x * blockDim.x + threadIdx.x;
  if (i < total) msum[i] /= fmaxf(counts[i >> 7], 1.0f);
}

// sub = z - mean_scale*mean[g]; vsum[g] += sub^2   (float4 lanes)
__global__ void node_sub(const float* __restrict__ z, const float* __restrict__ mean,
    const float* __restrict__ ga, const int* __restrict__ batch,
    float* __restrict__ sub, float* __restrict__ vsum, int total4) {
  int idx = blockIdx.x * blockDim.x + threadIdx.x;
  if (idx >= total4) return;
  int n = idx >> 5, f4 = idx & 31;
  int g = batch[n];
  float4 zv = ((const float4*)z)[idx];
  float4 gav = ((const float4*)ga)[f4];
  const float4 mv = ((const float4*)mean)[(g << 5) + f4];
  float4 s;
  s.x = zv.x - gav.x * mv.x;
  s.y = zv.y - gav.y * mv.y;
  s.z = zv.z - gav.z * mv.z;
  s.w = zv.w - gav.w * mv.w;
  ((float4*)sub)[idx] = s;
  float* vs = vsum + (g << 7) + f4 * 4;
  atomAddF(vs + 0, s.x * s.x); atomAddF(vs + 1, s.y * s.y);
  atomAddF(vs + 2, s.z * s.z); atomAddF(vs + 3, s.w * s.w);
}

// istd = rsqrt(vsum/counts + eps) (in place)
__global__ void finish_istd(float* __restrict__ vsum, const float* __restrict__ counts, int total) {
  int i = blockIdx.x * blockDim.x + threadIdx.x;
  if (i < total) vsum[i] = rsqrtf(vsum[i] / fmaxf(counts[i >> 7], 1.0f) + GN_EPS);
}

// out = gw*sub*istd[g] + gb (in place over sub)   (float4 lanes)
__global__ void norm_final(float* __restrict__ sub, const float* __restrict__ istd,
    const float* __restrict__ gw, const float* __restrict__ gb,
    const int* __restrict__ batch, int total4) {
  int idx = blockIdx.x * blockDim.x + threadIdx.x;
  if (idx >= total4) return;
  int n = idx >> 5, f4 = idx & 31;
  int g = batch[n];
  float4 s  = ((const float4*)sub)[idx];
  float4 wv = ((const float4*)gw)[f4];
  float4 bv = ((const float4*)gb)[f4];
  float4 iv = ((const float4*)istd)[(g << 5) + f4];
  s.x = wv.x * s.x * iv.x + bv.x;
  s.y = wv.y * s.y * iv.y + bv.y;
  s.z = wv.z * s.z * iv.z + bv.z;
  s.w = wv.w * s.w * iv.w + bv.w;
  ((float4*)sub)[idx] = s;
}

// layer-2 tail fused: o = gw*sub*istd[g] + gb; pooled[g] += o  (no N*H writeback)
__global__ void norm_pool(const float* __restrict__ sub, const float* __restrict__ istd,
    const float* __restrict__ gw, const float* __restrict__ gb,
    const int* __restrict__ batch, float* __restrict__ pooled, int total4) {
  int idx = blockIdx.x * blockDim.x + threadIdx.x;
  if (idx >= total4) return;
  int n = idx >> 5, f4 = idx & 31;
  int g = batch[n];
  float4 s  = ((const float4*)sub)[idx];
  float4 wv = ((const float4*)gw)[f4];
  float4 bv = ((const float4*)gb)[f4];
  float4 iv = ((const float4*)istd)[(g << 5) + f4];
  float* pp = pooled + (g << 7) + f4 * 4;
  atomAddF(pp + 0, wv.x * s.x * iv.x + bv.x);
  atomAddF(pp + 1, wv.y * s.y * iv.y + bv.y);
  atomAddF(pp + 2, wv.z * s.z * iv.z + bv.z);
  atomAddF(pp + 3, wv.w * s.w * iv.w + bv.w);
}

// logits = (pooled/cnt) @ Wc + bc ; softmax ; one wave per graph
__global__ __launch_bounds__(32) void classifier(const float* __restrict__ pooled,
    const float* __restrict__ counts, const float* __restrict__ Wc,
    const float* __restrict__ bc, float* __restrict__ out) {
  __shared__ float lg[NCLS];
  __shared__ float mx, sm;
  const int g = blockIdx.x, c = threadIdx.x;
  const float cinv = 1.0f / fmaxf(counts[g], 1.0f);
  if (c < NCLS) {
    float acc = bc[c];
    #pragma unroll 4
    for (int k = 0; k < HDIM; ++k)
      acc += (pooled[(g << 7) + k] * cinv) * Wc[k * NCLS + c];
    lg[c] = acc;
  }
  __syncthreads();
  if (c == 0) {
    float m = lg[0];
    for (int i = 1; i < NCLS; ++i) m = fmaxf(m, lg[i]);
    float s = 0.f;
    for (int i = 0; i < NCLS; ++i) s += __expf(lg[i] - m);
    mx = m; sm = s;
  }
  __syncthreads();
  if (c < NCLS) out[g * NCLS + c] = __expf(lg[c] - mx) / sm;
}

// ---------------- host orchestration ----------------
extern "C" void kernel_launch(void* const* d_in, const int* in_sizes, int n_in,
                              void* d_out, int out_size, void* d_ws, size_t ws_size,
                              hipStream_t stream) {
  // input order: x, edge_index, batch, W1,b1,gw1,gb1,ga1, W2,b2,gw2,gb2,ga2, Wc,bc
  const float* x   = (const float*)d_in[0];
  const int*   ei  = (const int*)  d_in[1];   // [2,E] flat: row = ei[0:E), col = ei[E:2E)
  const int*   bat = (const int*)  d_in[2];
  const float* W1  = (const float*)d_in[3];
  const float* b1  = (const float*)d_in[4];
  const float* gw1 = (const float*)d_in[5];
  const float* gb1 = (const float*)d_in[6];
  const float* ga1 = (const float*)d_in[7];
  const float* W2  = (const float*)d_in[8];
  const float* b2  = (const float*)d_in[9];
  const float* gw2 = (const float*)d_in[10];
  const float* gb2 = (const float*)d_in[11];
  const float* ga2 = (const float*)d_in[12];
  const float* Wc  = (const float*)d_in[13];
  const float* bc  = (const float*)d_in[14];
  float* out = (float*)d_out;

  const int   N = NNODES, E = NEDGES;
  const size_t NH  = (size_t)N * HDIM;
  const size_t NH4 = NH * sizeof(float);
  const size_t GH4 = (size_t)NGRAPH * HDIM * sizeof(float);

  // workspace carve-out
  char* base = (char*)d_ws;
  size_t off = 0;
  auto carve = [&](size_t bytes) -> char* {
    char* p = base + off;
    off = (off + bytes + 511) & ~(size_t)511;
    return p;
  };
  float*  bufA   = (float*)carve(NH4);                 // h / z (per layer)
  float*  bufB   = (float*)carve(NH4);                 // agg / sub / layer output
  float*  dis    = (float*)carve((size_t)N * 4);       // deg -> rsqrt(deg)
  float*  counts = (float*)carve((size_t)NGRAPH * 4);
  float*  msum   = (float*)carve(GH4);                 // mean accumulator -> mean
  float*  vsum   = (float*)carve(GH4);                 // var accumulator -> istd
  float*  pooled = (float*)carve(GH4);
  __bf16* W1s    = (__bf16*)carve((size_t)HDIM * HDIM * 2);
  __bf16* W2s    = (__bf16*)carve((size_t)HDIM * HDIM * 2);
  (void)ws_size;

  const int* erow = ei;
  const int* ecol = ei + E;

  const int TB = 256;
  const int nh4Blocks  = (int)((NH / 4 + TB - 1) / TB);
  const int ghBlocks   = (int)((NGRAPH * HDIM + TB - 1) / TB);
  const int gemmBlocks = (N + 127) / 128;
  const int edgeBlocks = (E + 7) / 8;

  // ---- graph structure prep ----
  swizzle_w<<<(HDIM * HDIM + TB - 1) / TB, TB, 0, stream>>>(W1, W1s);
  swizzle_w<<<(HDIM * HDIM + TB - 1) / TB, TB, 0, stream>>>(W2, W2s);
  fill_f32<<<(N + TB - 1) / TB, TB, 0, stream>>>(dis, 1.0f, N);      // self loop deg
  deg_edges<<<(E + TB - 1) / TB, TB, 0, stream>>>(ecol, dis, E);
  deg_to_rsqrt<<<(N + TB - 1) / TB, TB, 0, stream>>>(dis, N);
  hipMemsetAsync(counts, 0, (size_t)NGRAPH * 4, stream);
  count_nodes<<<(N + TB - 1) / TB, TB, 0, stream>>>(bat, counts, N);

  // ---- layer 1 (input x) ----
  gemm128_wmma<<<gemmBlocks, TB, 0, stream>>>(x, W1s, bufA, N);      // h1 -> bufA
  hipMemsetAsync(bufB, 0, NH4, stream);
  edge_scatter<<<edgeBlocks, TB, 0, stream>>>(erow, ecol, dis, bufA, bufB, E);
  hipMemsetAsync(msum, 0, GH4, stream);
  node_epilogue<<<nh4Blocks, TB, 0, stream>>>(bufA, bufB, dis, b1, bat, msum, (int)(NH / 4));
  finish_mean<<<ghBlocks, TB, 0, stream>>>(msum, counts, NGRAPH * HDIM);
  hipMemsetAsync(vsum, 0, GH4, stream);
  node_sub<<<nh4Blocks, TB, 0, stream>>>(bufA, msum, ga1, bat, bufB, vsum, (int)(NH / 4));
  finish_istd<<<ghBlocks, TB, 0, stream>>>(vsum, counts, NGRAPH * HDIM);
  norm_final<<<nh4Blocks, TB, 0, stream>>>(bufB, vsum, gw1, gb1, bat, (int)(NH / 4));
  // layer-1 output now in bufB

  // ---- layer 2 (input bufB) ----
  gemm128_wmma<<<gemmBlocks, TB, 0, stream>>>(bufB, W2s, bufA, N);   // h2 -> bufA
  hipMemsetAsync(bufB, 0, NH4, stream);                              // bufB dead after gemm
  edge_scatter<<<edgeBlocks, TB, 0, stream>>>(erow, ecol, dis, bufA, bufB, E);
  hipMemsetAsync(msum, 0, GH4, stream);
  node_epilogue<<<nh4Blocks, TB, 0, stream>>>(bufA, bufB, dis, b2, bat, msum, (int)(NH / 4));
  finish_mean<<<ghBlocks, TB, 0, stream>>>(msum, counts, NGRAPH * HDIM);
  hipMemsetAsync(vsum, 0, GH4, stream);
  node_sub<<<nh4Blocks, TB, 0, stream>>>(bufA, msum, ga2, bat, bufB, vsum, (int)(NH / 4));
  finish_istd<<<ghBlocks, TB, 0, stream>>>(vsum, counts, NGRAPH * HDIM);

  // ---- fused norm + pool, then classify + softmax ----
  hipMemsetAsync(pooled, 0, GH4, stream);
  norm_pool<<<nh4Blocks, TB, 0, stream>>>(bufB, vsum, gw2, gb2, bat, pooled, (int)(NH / 4));
  classifier<<<NGRAPH, 32, 0, stream>>>(pooled, counts, Wc, bc, out);
  (void)out_size; (void)n_in; (void)in_sizes;
}
